// TransformerBlock_14010183320274
// MI455X (gfx1250) — matrix operand order repaired and checked by reading, not compiled, not run on hardware
//
#include <hip/hip_runtime.h>
#include <hip/hip_bf16.h>
#include <math.h>

typedef __attribute__((ext_vector_type(16))) _Float16 v16h;
typedef __attribute__((ext_vector_type(8)))  _Float16 v8h;
typedef __attribute__((ext_vector_type(8)))  float    v8f;

#define HW   9216      // 96*96
#define NB   2         // batch
#define NTOT 18432     // NB*HW
#define CCH  64

// ---------------------------------------------------------------------------
// WMMA fragment loaders (ISA 7.12.2 layouts, 16-bit operands, wave32)
// A (16xK=32): lane<16 -> row=lane, k {0..7,16..23}; lane>=16 -> row=lane-16, k {8..15,24..31}
// B (K=32x16): lane<16 -> col=lane, k {0..7,16..23}; lane>=16 -> col=lane-16, k {8..15,24..31}
// ---------------------------------------------------------------------------
__device__ __forceinline__ v16h frag_a(const _Float16* p, int ld, int lane) {
  const _Float16* r = p + (size_t)(lane & 15) * ld + ((lane & 16) ? 8 : 0);
  v8h lo = *(const v8h*)(r);
  v8h hi = *(const v8h*)(r + 16);
  return __builtin_shufflevector(lo, hi, 0,1,2,3,4,5,6,7,8,9,10,11,12,13,14,15);
}
__device__ __forceinline__ v16h frag_b(const _Float16* p, int ld, int lane) {
  const _Float16* r = p + (size_t)(lane & 15) * ld + ((lane & 16) ? 8 : 0);
  v8h lo = *(const v8h*)(r);
  v8h hi = *(const v8h*)(r + 16);
  return __builtin_shufflevector(lo, hi, 0,1,2,3,4,5,6,7,8,9,10,11,12,13,14,15);
}
__device__ __forceinline__ v8f wmma_f16(v16h a, v16h b, v8f c) {
  return __builtin_amdgcn_wmma_f32_16x16x32_f16(false, a, false, b, (short)0, c, false, false);
}

// Async copy 16B global -> LDS (gfx1250 GLOBAL_LOAD_ASYNC_TO_LDS_B128, ASYNCcnt)
__device__ __forceinline__ void async_cp16(void* lds, const void* g) {
  unsigned l = (unsigned)(uintptr_t)lds;   // low 32 bits of flat LDS addr = LDS offset
  asm volatile("global_load_async_to_lds_b128 %0, %1, off" :: "v"(l), "v"(g) : "memory");
}
__device__ __forceinline__ void wait_async0() {
  asm volatile("s_wait_asynccnt 0x0" ::: "memory");
}
__device__ __forceinline__ void wait_ds0() {
  asm volatile("s_wait_dscnt 0x0" ::: "memory");
}

// ---------------------------------------------------------------------------
// f32 -> f16 weight conversion
// ---------------------------------------------------------------------------
__global__ void cvt_f16_kernel(const float* __restrict__ s, _Float16* __restrict__ d, int n) {
  int t = blockIdx.x * blockDim.x + threadIdx.x;
  if (t < n) d[t] = (_Float16)s[t];
}

// ---------------------------------------------------------------------------
// LayerNorm over channels of [B,C,HW] f32 -> position-major f16 [n, 64]
// ---------------------------------------------------------------------------
__global__ void ln_pack_kernel(const float* __restrict__ x, const float* __restrict__ gm,
                               const float* __restrict__ bt, _Float16* __restrict__ out) {
  int t = blockIdx.x * blockDim.x + threadIdx.x;
  if (t >= NTOT) return;
  int b = t / HW, hw = t - b * HW;
  const float* p = x + (size_t)b * CCH * HW + hw;
  float s = 0.f, s2 = 0.f;
  #pragma unroll 8
  for (int c = 0; c < CCH; ++c) { float v = p[(size_t)c * HW]; s += v; s2 += v * v; }
  float mu  = s * (1.f / CCH);
  float var = s2 * (1.f / CCH) - mu * mu;
  float inv = rsqrtf(var + 1e-5f);
  _Float16* o = out + (size_t)t * CCH;
  #pragma unroll 8
  for (int c = 0; c < CCH; ++c)
    o[c] = (_Float16)((p[(size_t)c * HW] - mu) * inv * gm[c] + bt[c]);
}

// ---------------------------------------------------------------------------
// Generic N x 16(o-tile) GEMM:  acc[n,o] = sum_k A[n,k] * W[o,k] (+bias)
// MODE 0: f16 out row-major [n,64]     MODE 1: f16 out channel-major [b][o][hw]
// MODE 2: f32 out channel-major + residual (final output)
// ---------------------------------------------------------------------------
template<int KTILES, int MODE>
__global__ __launch_bounds__(256) void gemm_kernel(
    const _Float16* __restrict__ A, const _Float16* __restrict__ W,
    const float* __restrict__ bias, void* __restrict__ outp,
    const float* __restrict__ res) {
  const int ldk  = KTILES * 32;
  int wave = threadIdx.x >> 5, lane = threadIdx.x & 31;
  int n0 = (blockIdx.x * 8 + wave) * 16;
  int o0 = blockIdx.y * 16;
  const _Float16* a0 = A + (size_t)n0 * ldk;
  const _Float16* w0 = W + (size_t)o0 * ldk;
  v8f acc = {};
  #pragma unroll
  for (int kt = 0; kt < KTILES; ++kt) {
    v16h af = frag_a(a0 + kt * 32, ldk, lane);
    v16h bf = frag_b(w0 + kt * 32, ldk, lane);
    acc = wmma_f16(af, bf, acc);
  }
  int o     = o0 + (lane & 15);
  int mbase = (lane >> 4) * 8;
  float bo  = bias ? bias[o] : 0.f;
  if (MODE == 0) {
    _Float16* out = (_Float16*)outp;
    #pragma unroll
    for (int i = 0; i < 8; ++i)
      out[(size_t)(n0 + mbase + i) * CCH + o] = (_Float16)(acc[i] + bo);
  } else if (MODE == 1) {
    _Float16* out = (_Float16*)outp;
    int b = n0 / HW, hw0 = n0 - b * HW;
    #pragma unroll
    for (int i = 0; i < 8; ++i)
      out[(size_t)b * CCH * HW + (size_t)o * HW + hw0 + mbase + i] = (_Float16)(acc[i] + bo);
  } else {
    float* out = (float*)outp;
    int b = n0 / HW, hw0 = n0 - b * HW;
    #pragma unroll
    for (int i = 0; i < 8; ++i) {
      size_t idx = (size_t)b * CCH * HW + (size_t)o * HW + hw0 + mbase + i;
      out[idx] = res[idx] + acc[i] + bo;
    }
  }
}

// ---------------------------------------------------------------------------
// Flash attention: out[b,c,n] = x[b,c,n] + sum_m V[c,m] * softmax_m(Q[n]·K[m])
// 8 waves/block, 16 query rows per wave. K/V tiles (32 m-steps) are staged
// into LDS once per block with async copies (double-buffered, ASYNCcnt),
// shared by all 8 waves -> 8x less L2 traffic, DMA overlaps WMMA+softmax.
// ---------------------------------------------------------------------------
__global__ __launch_bounds__(256) void attn_kernel(
    const _Float16* __restrict__ Q, const _Float16* __restrict__ K,
    const _Float16* __restrict__ V, const float* __restrict__ xin,
    float* __restrict__ xout) {
  __shared__ alignas(128) _Float16 Kbuf[2][32][64];   // [buf][m][c]
  __shared__ alignas(128) _Float16 Vbuf[2][64][32];   // [buf][c][m]
  __shared__ alignas(32)  float    s_s[8][16 * 32];   // raw S tile per wave
  __shared__ alignas(32)  _Float16 s_p[8][16 * 32];   // P = exp(S-max), [n][m]
  int tid  = threadIdx.x;
  int wave = tid >> 5, lane = tid & 31;
  int b  = blockIdx.y;
  int n0 = blockIdx.x * 128 + wave * 16;              // query row base (within batch)

  const _Float16* Kb = K + (size_t)b * HW * CCH;      // [m][c] row-major
  const _Float16* Vb = V + (size_t)b * CCH * HW;      // [c][m] channel-major

  // staging assignment: 256 threads x 16B cover each 4KB tile
  int km = tid >> 3, kc = (tid & 7) * 8;              // K: row m (0..31), col chunk
  int vc = tid >> 2, vm = (tid & 3) * 8;              // V: row c (0..63), col chunk

  const _Float16* qb = Q + ((size_t)b * HW + n0) * CCH;
  v16h qlo = frag_a(qb,      CCH, lane);
  v16h qhi = frag_a(qb + 32, CCH, lane);

  v8f o0 = {}, o1 = {}, o2 = {}, o3 = {};
  float rmax = -3.0e38f, rsum = 0.f;                  // per-row state, lane<16 owns row=lane

  float*    Srow = &s_s[wave][0];
  _Float16* Prow = &s_p[wave][0];

  // prologue: stage tile 0
  async_cp16(&Kbuf[0][km][kc], Kb + (size_t)km * CCH + kc);
  async_cp16(&Vbuf[0][vc][vm], Vb + (size_t)vc * HW + vm);
  wait_async0();
  __syncthreads();

  int cur = 0;
  for (int m0 = 0; m0 < HW; m0 += 32, cur ^= 1) {
    // ---- S = Q K^T (16 x 32) from LDS-staged K ----
    const _Float16* Kc = &Kbuf[cur][0][0];
    v16h kf0lo = frag_b(Kc,                CCH, lane);
    v16h kf0hi = frag_b(Kc + 32,           CCH, lane);
    v16h kf1lo = frag_b(Kc + 16 * CCH,     CCH, lane);
    v16h kf1hi = frag_b(Kc + 16 * CCH + 32, CCH, lane);
    v8f s0 = {}, s1 = {};
    s0 = wmma_f16(qlo, kf0lo, s0);
    s0 = wmma_f16(qhi, kf0hi, s0);
    s1 = wmma_f16(qlo, kf1lo, s1);
    s1 = wmma_f16(qhi, kf1hi, s1);

    // ---- stage S to LDS as [row n][col m] f32 ----
    int colb = lane & 15, rb = (lane >> 4) * 8;
    #pragma unroll
    for (int i = 0; i < 8; ++i) {
      Srow[(rb + i) * 32 + colb]      = s0[i];
      Srow[(rb + i) * 32 + colb + 16] = s1[i];
    }
    wait_ds0();
    __builtin_amdgcn_wave_barrier();

    // ---- kick off async DMA for the next tile (overlaps softmax + PV) ----
    int mn = m0 + 32;
    if (mn < HW) {
      async_cp16(&Kbuf[cur ^ 1][km][kc], Kb + (size_t)(mn + km) * CCH + kc);
      async_cp16(&Vbuf[cur ^ 1][vc][vm], Vb + (size_t)vc * HW + mn + vm);
    }

    // ---- online softmax: lane n owns row n ----
    float alpha = 1.f;
    if (lane < 16) {
      const float* row = Srow + lane * 32;
      float tmax = row[0];
      #pragma unroll
      for (int j = 1; j < 32; ++j) tmax = fmaxf(tmax, row[j]);
      float nmax = fmaxf(rmax, tmax);
      alpha = __expf(rmax - nmax);
      float ps = 0.f;
      _Float16* prow = Prow + lane * 32;
      #pragma unroll
      for (int j = 0; j < 32; ++j) {
        float e = __expf(row[j] - nmax);
        ps += e;
        prow[j] = (_Float16)e;
      }
      rsum = rsum * alpha + ps;
      rmax = nmax;
    }
    alpha = __shfl(alpha, lane & 15, 32);             // broadcast alpha_n to both half-waves
    #pragma unroll
    for (int i = 0; i < 8; ++i) { o0[i] *= alpha; o1[i] *= alpha; o2[i] *= alpha; o3[i] *= alpha; }

    wait_ds0();
    __builtin_amdgcn_wave_barrier();

    // ---- O^T (64 x 16) += V(64x32) * P^T(32x16), V from LDS ----
    v16h pb = frag_b(Prow, 32, lane);
    o0 = wmma_f16(frag_a(&Vbuf[cur][ 0][0], 32, lane), pb, o0);
    o1 = wmma_f16(frag_a(&Vbuf[cur][16][0], 32, lane), pb, o1);
    o2 = wmma_f16(frag_a(&Vbuf[cur][32][0], 32, lane), pb, o2);
    o3 = wmma_f16(frag_a(&Vbuf[cur][48][0], 32, lane), pb, o3);

    // ---- hand off double buffer ----
    wait_async0();
    __syncthreads();
  }

  // ---- normalize + residual + store (channel-major f32) ----
  float inv = 1.f / rsum;
  inv = __shfl(inv, lane & 15, 32);
  int ncol = lane & 15, mrow = (lane >> 4) * 8;
  v8f* ot[4] = { &o0, &o1, &o2, &o3 };
  #pragma unroll
  for (int t = 0; t < 4; ++t)
    #pragma unroll
    for (int i = 0; i < 8; ++i) {
      int c = t * 16 + mrow + i;
      size_t idx = (size_t)b * CCH * HW + (size_t)c * HW + n0 + ncol;
      xout[idx] = xin[idx] + (*ot[t])[i] * inv;
    }
}

// ---------------------------------------------------------------------------
// 8x8 patch: irfft2(rfft2(h) * fft_w), one thread per (b, patch, channel)
// ---------------------------------------------------------------------------
__constant__ float c_cos8[8] = {1.f, 0.70710678f, 0.f, -0.70710678f, -1.f, -0.70710678f, 0.f, 0.70710678f};
__constant__ float c_sin8[8] = {0.f, 0.70710678f, 1.f, 0.70710678f, 0.f, -0.70710678f, -1.f, -0.70710678f};

__global__ void fft_filter_kernel(const _Float16* __restrict__ h0,
                                  const float* __restrict__ fw,
                                  float* __restrict__ h1) {
  int t = blockIdx.x * blockDim.x + threadIdx.x;
  if (t >= NTOT) return;
  int c = t & 63, p = t >> 6;
  int px = p % 12, py = (p / 12) % 12, b = p / 144;
  int base_hw = (py * 8) * 96 + px * 8;
  const _Float16* src = h0 + ((size_t)(b * HW + base_hw)) * CCH + c;

  // forward row rFFT (k = 0..4)
  float Ar[8][5], Ai[8][5];
  #pragma unroll
  for (int r = 0; r < 8; ++r) {
    float xv[8];
    #pragma unroll
    for (int col = 0; col < 8; ++col) xv[col] = (float)src[(size_t)(r * 96 + col) * CCH];
    #pragma unroll
    for (int kx = 0; kx < 5; ++kx) {
      float ar = 0.f, ai = 0.f;
      #pragma unroll
      for (int col = 0; col < 8; ++col) {
        int idx = (kx * col) & 7;
        ar += xv[col] * c_cos8[idx];
        ai -= xv[col] * c_sin8[idx];
      }
      Ar[r][kx] = ar; Ai[r][kx] = ai;
    }
  }
  float y[64];
  #pragma unroll
  for (int i = 0; i < 64; ++i) y[i] = 0.f;
  const float* fwc = fw + c * 40;
  for (int ky = 0; ky < 8; ++ky) {                   // rolled; kx unrolled keeps reg indices static
    #pragma unroll
    for (int kx = 0; kx < 5; ++kx) {
      float zr = 0.f, zi = 0.f;
      #pragma unroll
      for (int r = 0; r < 8; ++r) {
        int idx = (ky * r) & 7;
        float cc = c_cos8[idx], ss = c_sin8[idx];
        zr += Ar[r][kx] * cc + Ai[r][kx] * ss;
        zi += Ai[r][kx] * cc - Ar[r][kx] * ss;
      }
      float wv = fwc[ky * 5 + kx];
      zr *= wv; zi *= wv;
      float m = (kx == 0 || kx == 4) ? 1.f : 2.f;    // Hermitian half-spectrum weight
      #pragma unroll
      for (int r2 = 0; r2 < 8; ++r2)
        #pragma unroll
        for (int c2 = 0; c2 < 8; ++c2) {
          int idx = (ky * r2 + kx * c2) & 7;
          y[r2 * 8 + c2] += m * (zr * c_cos8[idx] - zi * c_sin8[idx]);
        }
    }
  }
  float* dst = h1 + (size_t)b * CCH * HW + (size_t)c * HW + base_hw;
  #pragma unroll
  for (int r2 = 0; r2 < 8; ++r2)
    #pragma unroll
    for (int c2 = 0; c2 < 8; ++c2)
      dst[r2 * 96 + c2] = y[r2 * 8 + c2] * (1.f / 64.f);
}

// ---------------------------------------------------------------------------
// depthwise 3x3 + exact-GELU gate -> g f16 position-major [n, 32]
// ---------------------------------------------------------------------------
__global__ void dwconv_gate_kernel(const float* __restrict__ h1,
                                   const float* __restrict__ dw,
                                   _Float16* __restrict__ g) {
  int t = blockIdx.x * blockDim.x + threadIdx.x;
  if (t >= NB * 32 * HW) return;
  int n = t % HW, c = (t / HW) & 31, b = t / (HW * 32);
  int h = n / 96, w = n % 96;
  const float* p0 = h1 + (size_t)b * CCH * HW + (size_t)c * HW;
  const float* p1 = p0 + (size_t)32 * HW;
  const float* w0 = dw + c * 9;
  const float* w1 = dw + (c + 32) * 9;
  float a0 = 0.f, a1 = 0.f;
  #pragma unroll
  for (int dy = -1; dy <= 1; ++dy) {
    int hy = h + dy;
    if (hy < 0 || hy >= 96) continue;
    #pragma unroll
    for (int dx = -1; dx <= 1; ++dx) {
      int wx = w + dx;
      if (wx < 0 || wx >= 96) continue;
      int off = hy * 96 + wx, wi = (dy + 1) * 3 + (dx + 1);
      a0 += p0[off] * w0[wi];
      a1 += p1[off] * w1[wi];
    }
  }
  float ge = 0.5f * a0 * (1.f + erff(a0 * 0.70710678f));
  g[(size_t)(b * HW + n) * 32 + c] = (_Float16)(ge * a1);
}

// ---------------------------------------------------------------------------
// launch
// ---------------------------------------------------------------------------
extern "C" void kernel_launch(void* const* d_in, const int* in_sizes, int n_in,
                              void* d_out, int out_size, void* d_ws, size_t ws_size,
                              hipStream_t stream) {
  (void)in_sizes; (void)n_in; (void)out_size; (void)ws_size;
  const float* x      = (const float*)d_in[0];
  const float* ln1_w  = (const float*)d_in[1];
  const float* ln1_b  = (const float*)d_in[2];
  const float* ln2_w  = (const float*)d_in[3];
  const float* ln2_b  = (const float*)d_in[4];
  const float* q_w    = (const float*)d_in[5];
  const float* q_b    = (const float*)d_in[6];
  const float* k_w    = (const float*)d_in[7];
  const float* k_b    = (const float*)d_in[8];
  const float* v_w    = (const float*)d_in[9];
  const float* v_b    = (const float*)d_in[10];
  const float* fft_w  = (const float*)d_in[11];
  const float* pin_w  = (const float*)d_in[12];
  const float* dw_w   = (const float*)d_in[13];
  const float* pout_w = (const float*)d_in[14];

  char* w = (char*)d_ws;
  auto alloc = [&](size_t bytes) { char* p = w; w += (bytes + 255) & ~(size_t)255; return p; };
  _Float16* xn1   = (_Float16*)alloc((size_t)NTOT * CCH * 2);
  _Float16* Qf    = (_Float16*)alloc((size_t)NTOT * CCH * 2);
  _Float16* Kf    = (_Float16*)alloc((size_t)NTOT * CCH * 2);
  _Float16* Vf    = (_Float16*)alloc((size_t)NTOT * CCH * 2);   // channel-major per batch
  float*    xatt  = (float*)   alloc((size_t)NTOT * CCH * 4);
  _Float16* xn2   = (_Float16*)alloc((size_t)NTOT * CCH * 2);
  _Float16* h0    = (_Float16*)alloc((size_t)NTOT * CCH * 2);
  float*    h1    = (float*)   alloc((size_t)NTOT * CCH * 4);
  _Float16* gbuf  = (_Float16*)alloc((size_t)NTOT * 32 * 2);
  _Float16* wq16  = (_Float16*)alloc(4096 * 2);
  _Float16* wk16  = (_Float16*)alloc(4096 * 2);
  _Float16* wv16  = (_Float16*)alloc(4096 * 2);
  _Float16* wpi16 = (_Float16*)alloc(4096 * 2);
  _Float16* wpo16 = (_Float16*)alloc(2048 * 2);

  cvt_f16_kernel<<<16, 256, 0, stream>>>(q_w,    wq16,  4096);
  cvt_f16_kernel<<<16, 256, 0, stream>>>(k_w,    wk16,  4096);
  cvt_f16_kernel<<<16, 256, 0, stream>>>(v_w,    wv16,  4096);
  cvt_f16_kernel<<<16, 256, 0, stream>>>(pin_w,  wpi16, 4096);
  cvt_f16_kernel<<<8,  256, 0, stream>>>(pout_w, wpo16, 2048);

  ln_pack_kernel<<<72, 256, 0, stream>>>(x, ln1_w, ln1_b, xn1);

  dim3 ggrid(144, 4);
  gemm_kernel<2, 0><<<ggrid, 256, 0, stream>>>(xn1, wq16, q_b, Qf, nullptr);
  gemm_kernel<2, 0><<<ggrid, 256, 0, stream>>>(xn1, wk16, k_b, Kf, nullptr);
  gemm_kernel<2, 1><<<ggrid, 256, 0, stream>>>(xn1, wv16, v_b, Vf, nullptr);

  attn_kernel<<<dim3(72, 2), 256, 0, stream>>>(Qf, Kf, Vf, x, xatt);

  ln_pack_kernel<<<72, 256, 0, stream>>>(xatt, ln2_w, ln2_b, xn2);
  gemm_kernel<2, 0><<<ggrid, 256, 0, stream>>>(xn2, wpi16, nullptr, h0, nullptr);

  fft_filter_kernel<<<72, 256, 0, stream>>>(h0, fft_w, h1);
  dwconv_gate_kernel<<<2304, 256, 0, stream>>>(h1, dw_w, gbuf);

  gemm_kernel<1, 2><<<ggrid, 256, 0, stream>>>(gbuf, wpo16, nullptr, d_out, xatt);
}
